// SlotAttentionModule_51505247814182
// MI455X (gfx1250) — matrix operand order repaired and checked by reading, not compiled, not run on hardware
//
#include <hip/hip_runtime.h>
#include <hip/hip_bf16.h>

typedef __attribute__((ext_vector_type(16))) _Float16 v16h;
typedef __attribute__((ext_vector_type(8)))  _Float16 v8h;
typedef __attribute__((ext_vector_type(8)))  float    v8f;

#define NB 64
#define NN 4096
#define NC 256
#define ND 256
#define NS 8
#define NH 512
#define SCALE 0.0625f
#define LN_EPS 1e-3f
#define ATT_EPS 1e-8f

// ---------------- WMMA fragment loaders (CDNA5 16x16x32 f16) ----------------
// A (16x32, 16-bit): lane m = lane&15, half h = lane>>4.
//   elems 0..7  -> K = 8h + e ; elems 8..15 -> K = 16 + 8h + e
__device__ __forceinline__ v16h load_A(const _Float16* Ap, int lda, int lane) {
    int mm = lane & 15, h = lane >> 4;
    const _Float16* p = Ap + (size_t)mm * lda + 8 * h;
    v8h lo = *(const v8h*)(p);
    v8h hi = *(const v8h*)(p + 16);
    v16h a;
#pragma unroll
    for (int e = 0; e < 8; ++e) { a[e] = lo[e]; a[8 + e] = hi[e]; }
    return a;
}
// B (32x16, 16-bit): lane n = lane&15, half h = lane>>4; elems 0..7 -> K = 8h + e ; elems 8..15 -> K = 16 + 8h + e. Bp at (k0,n0), row-major [K x N].
__device__ __forceinline__ v16h load_B(const _Float16* Bp, int ldb, int lane) {
    int nn = lane & 15, h = lane >> 4;
    const _Float16* p = Bp + nn + (size_t)(8 * h) * ldb;
    v16h b;
#pragma unroll
    for (int e = 0; e < 8; ++e) { b[e] = p[(size_t)e * ldb]; b[8 + e] = p[(size_t)(16 + e) * ldb]; }
    return b;
}
__device__ __forceinline__ v8f wmma_f16(v16h a, v16h b, v8f c) {
    return __builtin_amdgcn_wmma_f32_16x16x32_f16(false, a, false, b, (short)0, c, false, false);
}
// attn (f32) -> A fragment with per-row normalization; rows >= NS are zero padding
__device__ __forceinline__ v16h load_attnA(const float* p, float sc, bool valid) {
    v16h a = {};
    if (valid) {
#pragma unroll
        for (int e = 0; e < 8; ++e) {
            a[e]     = (_Float16)(p[e] * sc);
            a[8 + e] = (_Float16)(p[16 + e] * sc);
        }
    }
    return a;
}

// ---------------- weight prep: transpose f32 [R,Cl] -> f16 [Cl,R] ----------------
__global__ void transpose_to_f16(const float* __restrict__ src, _Float16* __restrict__ dst,
                                 int R, int Cl) {
    int idx = blockIdx.x * 256 + threadIdx.x;
    if (idx >= R * Cl) return;
    int r = idx / Cl, c = idx % Cl;
    dst[(size_t)c * R + r] = (_Float16)src[idx];
}

// slots = mu + exp(log_sigma)*noise ; also zero padded q buffer [B,16,D]
__global__ void init_slots(const float* __restrict__ mu, const float* __restrict__ ls,
                           const float* __restrict__ noise, float* __restrict__ slots,
                           _Float16* __restrict__ qf) {
    int idx = blockIdx.x * 256 + threadIdx.x;   // 131072
    int d = idx & (ND - 1);
    slots[idx] = mu[d] + expf(ls[d]) * noise[idx];
    qf[idx] = (_Float16)0.0f;
    qf[idx + NB * NS * ND] = (_Float16)0.0f;
}

// ---------------- fused LN(inputs) + k/v projection (WMMA) ----------------
// block = 256 threads (8 waves), strip of 32 rows. k stored transposed [B,D,N], v row-major [B,N,D].
__global__ __launch_bounds__(256) void lnkv_kernel(
    const float* __restrict__ x, const float* __restrict__ gw, const float* __restrict__ bw,
    const _Float16* __restrict__ WkT, const _Float16* __restrict__ WvT,
    _Float16* __restrict__ kT, _Float16* __restrict__ vmat) {
    __shared__ float    xs[32 * NC];
    __shared__ _Float16 xh[32 * NC];
    int g0 = blockIdx.x * 32;
    int b  = g0 >> 12;
    int n0 = g0 & (NN - 1);
    const float* src = x + (size_t)g0 * NC;
    for (int i = threadIdx.x; i < 32 * NC; i += 256) xs[i] = src[i];
    __syncthreads();

    int lane = threadIdx.x & 31, wv = threadIdx.x >> 5;
    for (int rr = 0; rr < 4; ++rr) {               // LayerNorm: wave handles 4 rows
        int row = wv * 4 + rr;
        const float* xr = xs + row * NC;
        float s = 0.f;
        for (int i = lane; i < NC; i += 32) s += xr[i];
#pragma unroll
        for (int off = 16; off; off >>= 1) s += __shfl_xor(s, off, 32);
        float mu = s * (1.0f / NC);
        float vs = 0.f;
        for (int i = lane; i < NC; i += 32) { float d = xr[i] - mu; vs += d * d; }
#pragma unroll
        for (int off = 16; off; off >>= 1) vs += __shfl_xor(vs, off, 32);
        float rs = rsqrtf(vs * (1.0f / NC) + LN_EPS);
        for (int i = lane; i < NC; i += 32)
            xh[row * NC + i] = (_Float16)((xr[i] - mu) * rs * gw[i] + bw[i]);
    }
    __syncthreads();

    for (int t = 0; t < 8; ++t) {                  // 64 tiles / 8 waves
        int id = wv * 8 + t;
        int sel = id >> 5, rem = id & 31, mt = rem >> 4, nt = rem & 15;
        const _Float16* Bt = (sel ? WvT : WkT) + nt * 16;
        const _Float16* Ap = xh + mt * 16 * NC;
        v16h breg[8];                              // batch-preload all B fragments
#pragma unroll
        for (int kk = 0; kk < 8; ++kk)
            breg[kk] = load_B(Bt + (size_t)kk * 32 * ND, ND, lane);
        v8f c = {};
#pragma unroll
        for (int kk = 0; kk < 8; ++kk) {           // A from LDS interleaved with WMMA chain
            v16h a = load_A(Ap + kk * 32, NC, lane);
            c = wmma_f16(a, breg[kk], c);
        }
        int lcol = lane & 15, h = lane >> 4;
        int d = nt * 16 + lcol;
        if (sel == 0) {
            int nb = n0 + mt * 16 + 8 * h;
            v8h pk;
#pragma unroll
            for (int r = 0; r < 8; ++r) pk[r] = (_Float16)c[r];
            *(v8h*)(kT + ((size_t)b * ND + d) * NN + nb) = pk;
        } else {
#pragma unroll
            for (int r = 0; r < 8; ++r) {
                int nn = n0 + mt * 16 + r + 8 * h;
                vmat[((size_t)b * NN + nn) * ND + d] = (_Float16)c[r];
            }
        }
    }
}

// LN rows of [512,256] f32 -> f16 (used for slot-LN and pre-MLP LN)
__global__ __launch_bounds__(256) void ln_rows_f16(
    const float* __restrict__ src, const float* __restrict__ gw, const float* __restrict__ bw,
    _Float16* __restrict__ dst) {
    int lane = threadIdx.x & 31, wv = threadIdx.x >> 5;
    int row = blockIdx.x * 8 + wv;                 // 512 rows
    const float* p = src + (size_t)row * ND;
    float s = 0.f;
    for (int i = lane; i < ND; i += 32) s += p[i];
#pragma unroll
    for (int off = 16; off; off >>= 1) s += __shfl_xor(s, off, 32);
    float mu = s * (1.0f / ND);
    float vs = 0.f;
    for (int i = lane; i < ND; i += 32) { float d = p[i] - mu; vs += d * d; }
#pragma unroll
    for (int off = 16; off; off >>= 1) vs += __shfl_xor(vs, off, 32);
    float rs = rsqrtf(vs * (1.0f / ND) + LN_EPS);
    for (int i = lane; i < ND; i += 32)
        dst[(size_t)row * ND + i] = (_Float16)((p[i] - mu) * rs * gw[i] + bw[i]);
}

// ---------------- generic small WMMA GEMM, one 16x16 tile per wave ----------------
// MODE 0: f32 out [M,Nc] | 1: q padded f16 [B,16,D] | 2: +bias, ReLU, f16 out | 3: +bias + residual into f32
template <int MODE>
__global__ __launch_bounds__(256) void gemm16(
    const _Float16* __restrict__ A, const _Float16* __restrict__ Bm,
    float* __restrict__ outF, _Float16* __restrict__ outH,
    const float* __restrict__ bias, float* __restrict__ resid,
    int M, int K, int Nc) {
    int lane = threadIdx.x & 31, wv = threadIdx.x >> 5;
    int ntiles = Nc >> 4;
    int tile = blockIdx.x * 8 + wv;
    int mt = tile / ntiles, nt = tile % ntiles;
    int m0 = mt * 16, n0 = nt * 16;
    const _Float16* Arow = A + (size_t)m0 * K;
    const _Float16* Bt   = Bm + n0;
    v8f c = {};
    for (int k0 = 0; k0 < K; k0 += 32) {
        v16h a = load_A(Arow + k0, K, lane);
        v16h b = load_B(Bt + (size_t)k0 * Nc, Nc, lane);
        c = wmma_f16(a, b, c);
    }
    int lcol = lane & 15, h = lane >> 4;
#pragma unroll
    for (int r = 0; r < 8; ++r) {
        int row = m0 + r + 8 * h;
        int col = n0 + lcol;
        if (MODE == 0) {
            outF[(size_t)row * Nc + col] = c[r];
        } else if (MODE == 1) {
            int bi = row >> 3, s = row & 7;        // scatter into padded [B,16,D]
            outH[((size_t)bi * 16 + s) * ND + col] = (_Float16)c[r];
        } else if (MODE == 2) {
            float v = c[r] + bias[col];
            outH[(size_t)row * Nc + col] = (_Float16)(v > 0.f ? v : 0.f);
        } else {
            size_t idx = (size_t)row * Nc + col;
            resid[idx] = resid[idx] + c[r] + bias[col];
        }
    }
}

// dots[b,s,n] = SCALE * q @ k^T ; q padded [B,16,D], kT [B,D,N]
__global__ __launch_bounds__(256) void dots_kernel(
    const _Float16* __restrict__ qf, const _Float16* __restrict__ kT, float* __restrict__ dots) {
    int lane = threadIdx.x & 31, wv = threadIdx.x >> 5;
    int tile = blockIdx.x * 8 + wv;                // 16384 tiles
    int b = tile >> 8, nt = tile & 255;
    const _Float16* Ap = qf + (size_t)b * 16 * ND;
    const _Float16* Bp = kT + (size_t)b * ND * NN + nt * 16;
    v16h areg[8], breg[8];                         // batch-preload: loads issue as one clause,
#pragma unroll                                     // WMMA chain drains with partial waits
    for (int kk = 0; kk < 8; ++kk) areg[kk] = load_A(Ap + kk * 32, ND, lane);
#pragma unroll
    for (int kk = 0; kk < 8; ++kk) breg[kk] = load_B(Bp + (size_t)kk * 32 * NN, NN, lane);
    v8f c = {};
#pragma unroll
    for (int kk = 0; kk < 8; ++kk) c = wmma_f16(areg[kk], breg[kk], c);
    int lcol = lane & 15, h = lane >> 4;
    if (h == 0) {                                  // rows s=0..7 valid, 8..15 are padding
#pragma unroll
        for (int r = 0; r < 8; ++r)
            dots[(size_t)b * NS * NN + (size_t)r * NN + nt * 16 + lcol] = c[r] * SCALE;
    }
}

// softmax over the SLOT axis (8 values, strided) per (b,n); +EPS, in place
__global__ void softmax_slots(float* __restrict__ dots) {
    int idx = blockIdx.x * 256 + threadIdx.x;      // 262144
    int b = idx >> 12, n = idx & (NN - 1);
    float* p = dots + (size_t)b * NS * NN + n;
    float v[NS], mx = -1e30f;
#pragma unroll
    for (int s = 0; s < NS; ++s) { v[s] = p[(size_t)s * NN]; mx = fmaxf(mx, v[s]); }
    float sum = 0.f;
#pragma unroll
    for (int s = 0; s < NS; ++s) { v[s] = expf(v[s] - mx); sum += v[s]; }
    float inv = 1.0f / sum;
#pragma unroll
    for (int s = 0; s < NS; ++s) p[(size_t)s * NN] = v[s] * inv + ATT_EPS;
}

// reciprocal row sums over N per (b,s)
__global__ __launch_bounds__(256) void denom_kernel(const float* __restrict__ attn,
                                                    float* __restrict__ rden) {
    int lane = threadIdx.x & 31, wv = threadIdx.x >> 5;
    int row = blockIdx.x * 8 + wv;                 // 512 rows of [B*S, N]
    const float* p = attn + (size_t)row * NN;
    float s = 0.f;
    for (int i = lane; i < NN; i += 32) s += p[i];
#pragma unroll
    for (int off = 16; off; off >>= 1) s += __shfl_xor(s, off, 32);
    if (lane == 0) rden[row] = 1.0f / s;
}

// updates[b,s,d] = (attn/denom) @ v ; attn f32 [B,S,N] normalized in the A loader
__global__ __launch_bounds__(256) void updates_kernel(
    const float* __restrict__ attn, const float* __restrict__ rden,
    const _Float16* __restrict__ vmat, _Float16* __restrict__ upf) {
    int lane = threadIdx.x & 31, wv = threadIdx.x >> 5;
    int tile = blockIdx.x * 8 + wv;                // 1024 tiles
    int b = tile >> 4, dt = tile & 15;
    int mm = lane & 15, h = lane >> 4;
    bool valid = (mm < NS);
    float sc = 0.f;
    const float* pA = attn;
    if (valid) {
        sc = rden[b * NS + mm];
        pA = attn + (size_t)b * NS * NN + (size_t)mm * NN + 8 * h;
    }
    const _Float16* Bp = vmat + (size_t)b * NN * ND + dt * 16;
    v8f c = {};
#pragma unroll 2
    for (int k0 = 0; k0 < NN; k0 += 32) {
        v16h a = load_attnA(pA + k0, sc, valid);
        v16h bm = load_B(Bp + (size_t)k0 * ND, ND, lane);
        c = wmma_f16(a, bm, c);
    }
    int lcol = lane & 15;
    if (h == 0) {
#pragma unroll
        for (int r = 0; r < 8; ++r)
            upf[((size_t)b * NS + r) * ND + dt * 16 + lcol] = (_Float16)c[r];
    }
}

__global__ void f32_to_f16(const float* __restrict__ src, _Float16* __restrict__ dst, int n) {
    int i = blockIdx.x * 256 + threadIdx.x;
    if (i < n) dst[i] = (_Float16)src[i];
}

// GRU gates: r=sig(xr+hr), z=sig(xz+hz), n=tanh(xn+r*hn); slots=(1-z)*n+z*prev
__global__ void gru_gate(const float* __restrict__ gx, const float* __restrict__ gh,
                         const float* __restrict__ bih, const float* __restrict__ bhh,
                         float* __restrict__ slots) {
    int idx = blockIdx.x * 256 + threadIdx.x;      // 131072
    int row = idx >> 8, d = idx & (ND - 1);
    const float* gxr = gx + (size_t)row * 3 * ND;
    const float* ghr = gh + (size_t)row * 3 * ND;
    float xr = gxr[d]          + bih[d],          hr = ghr[d]          + bhh[d];
    float xz = gxr[ND + d]     + bih[ND + d],     hz = ghr[ND + d]     + bhh[ND + d];
    float xn = gxr[2 * ND + d] + bih[2 * ND + d], hn = ghr[2 * ND + d] + bhh[2 * ND + d];
    float r = 1.0f / (1.0f + expf(-(xr + hr)));
    float z = 1.0f / (1.0f + expf(-(xz + hz)));
    float n = tanhf(xn + r * hn);
    float sp = slots[idx];
    slots[idx] = (1.0f - z) * n + z * sp;
}

__global__ void copy_out(const float* __restrict__ src, float* __restrict__ dst) {
    int i = blockIdx.x * 256 + threadIdx.x;
    dst[i] = src[i];
}

extern "C" void kernel_launch(void* const* d_in, const int* in_sizes, int n_in,
                              void* d_out, int out_size, void* d_ws, size_t ws_size,
                              hipStream_t stream) {
    const float* inputs   = (const float*)d_in[0];
    const float* noise    = (const float*)d_in[1];
    const float* slots_mu = (const float*)d_in[2];
    const float* slots_ls = (const float*)d_in[3];
    const float* Wq       = (const float*)d_in[4];
    const float* Wk       = (const float*)d_in[5];
    const float* Wv       = (const float*)d_in[6];
    const float* W_ih     = (const float*)d_in[7];
    const float* W_hh     = (const float*)d_in[8];
    const float* b_ih     = (const float*)d_in[9];
    const float* b_hh     = (const float*)d_in[10];
    const float* mlp_W1   = (const float*)d_in[11];
    const float* mlp_b1   = (const float*)d_in[12];
    const float* mlp_W2   = (const float*)d_in[13];
    const float* mlp_b2   = (const float*)d_in[14];
    const float* ln_in_g  = (const float*)d_in[15];
    const float* ln_in_b  = (const float*)d_in[16];
    const float* ln_s_g   = (const float*)d_in[17];
    const float* ln_s_b   = (const float*)d_in[18];
    const float* ln_ff_g  = (const float*)d_in[19];
    const float* ln_ff_b  = (const float*)d_in[20];

    char* ws = (char*)d_ws;
    size_t off = 0;
    auto alloc = [&](size_t bytes) -> char* {
        char* p = ws + off;
        off = (off + bytes + 255) & ~(size_t)255;
        return p;
    };
    _Float16* kT       = (_Float16*)alloc((size_t)NB * ND * NN * 2);
    _Float16* vmat     = (_Float16*)alloc((size_t)NB * NN * ND * 2);
    float*    dots     = (float*)   alloc((size_t)NB * NS * NN * 4);
    float*    rden     = (float*)   alloc((size_t)NB * NS * 4);
    float*    slots    = (float*)   alloc((size_t)NB * NS * ND * 4);
    _Float16* upf16    = (_Float16*)alloc((size_t)NB * NS * ND * 2);
    _Float16* slotsf16 = (_Float16*)alloc((size_t)NB * NS * ND * 2);
    _Float16* slnf16   = (_Float16*)alloc((size_t)NB * NS * ND * 2);
    _Float16* qf16     = (_Float16*)alloc((size_t)NB * 16 * ND * 2);
    float*    gx       = (float*)   alloc((size_t)NB * NS * 3 * ND * 4);
    float*    gh       = (float*)   alloc((size_t)NB * NS * 3 * ND * 4);
    _Float16* pf16     = (_Float16*)alloc((size_t)NB * NS * ND * 2);
    _Float16* h1f16    = (_Float16*)alloc((size_t)NB * NS * NH * 2);
    _Float16* WkT      = (_Float16*)alloc((size_t)NC * ND * 2);
    _Float16* WvT      = (_Float16*)alloc((size_t)NC * ND * 2);
    _Float16* WqT      = (_Float16*)alloc((size_t)ND * ND * 2);
    _Float16* WihT     = (_Float16*)alloc((size_t)ND * 3 * ND * 2);
    _Float16* WhhT     = (_Float16*)alloc((size_t)ND * 3 * ND * 2);
    _Float16* W1T      = (_Float16*)alloc((size_t)ND * NH * 2);
    _Float16* W2T      = (_Float16*)alloc((size_t)NH * ND * 2);

    // weight prep (f32 -> transposed f16 [K,N] layouts)
    transpose_to_f16<<<256, 256, 0, stream>>>(Wk, WkT, ND, NC);
    transpose_to_f16<<<256, 256, 0, stream>>>(Wv, WvT, ND, NC);
    transpose_to_f16<<<256, 256, 0, stream>>>(Wq, WqT, ND, ND);
    transpose_to_f16<<<768, 256, 0, stream>>>(W_ih, WihT, 3 * ND, ND);
    transpose_to_f16<<<768, 256, 0, stream>>>(W_hh, WhhT, 3 * ND, ND);
    transpose_to_f16<<<512, 256, 0, stream>>>(mlp_W1, W1T, NH, ND);
    transpose_to_f16<<<512, 256, 0, stream>>>(mlp_W2, W2T, ND, NH);

    init_slots<<<512, 256, 0, stream>>>(slots_mu, slots_ls, noise, slots, qf16);
    lnkv_kernel<<<(NB * NN) / 32, 256, 0, stream>>>(inputs, ln_in_g, ln_in_b, WkT, WvT, kT, vmat);

    for (int it = 0; it < 3; ++it) {
        ln_rows_f16<<<64, 256, 0, stream>>>(slots, ln_s_g, ln_s_b, slnf16);
        gemm16<1><<<64, 256, 0, stream>>>(slnf16, WqT, nullptr, qf16, nullptr, nullptr,
                                          NB * NS, ND, ND);
        dots_kernel<<<2048, 256, 0, stream>>>(qf16, kT, dots);
        softmax_slots<<<1024, 256, 0, stream>>>(dots);
        denom_kernel<<<64, 256, 0, stream>>>(dots, rden);
        updates_kernel<<<128, 256, 0, stream>>>(dots, rden, vmat, upf16);
        f32_to_f16<<<512, 256, 0, stream>>>(slots, slotsf16, NB * NS * ND);
        gemm16<0><<<192, 256, 0, stream>>>(upf16, WihT, gx, nullptr, nullptr, nullptr,
                                           NB * NS, ND, 3 * ND);
        gemm16<0><<<192, 256, 0, stream>>>(slotsf16, WhhT, gh, nullptr, nullptr, nullptr,
                                           NB * NS, ND, 3 * ND);
        gru_gate<<<512, 256, 0, stream>>>(gx, gh, b_ih, b_hh, slots);
        ln_rows_f16<<<64, 256, 0, stream>>>(slots, ln_ff_g, ln_ff_b, pf16);
        gemm16<2><<<128, 256, 0, stream>>>(pf16, W1T, nullptr, h1f16, mlp_b1, nullptr,
                                           NB * NS, ND, NH);
        gemm16<3><<<64, 256, 0, stream>>>(h1f16, W2T, nullptr, nullptr, mlp_b2, slots,
                                          NB * NS, NH, ND);
    }
    copy_out<<<512, 256, 0, stream>>>(slots, (float*)d_out);
    (void)in_sizes; (void)n_in; (void)out_size; (void)ws_size;
}
